// MultiHeadSelfAttention_3083786519233
// MI455X (gfx1250) — matrix operand order repaired and checked by reading, not compiled, not run on hardware
//
#include <hip/hip_runtime.h>

// ---------------- types ----------------
typedef __bf16 bf16_t;
typedef __attribute__((ext_vector_type(16))) __bf16 bf16x16;
typedef __attribute__((ext_vector_type(4)))  __bf16 bf16x4;
typedef __attribute__((ext_vector_type(8)))  float  f32x8;

// ---------------- problem dims ----------------
#define NROWS   65536
#define F_DIM   1024
#define H_NUM   16
#define D_DIM   64
#define OUT_DIM 1024
#define QKV_N   3072           // 3 * H * D

// ---------------- tiling ----------------
#define TILE_ROWS 32
#define QKV_PITCH 3080         // bf16 elems per LDS row (3072 + 8 pad -> +16B shift/row)
#define X_PITCH   1032         // bf16 elems per LDS row (1024 + 8 pad)
#define LDS_QKV_BYTES (TILE_ROWS * QKV_PITCH * 2)   // 197120
#define LDS_X_BYTES   (TILE_ROWS * X_PITCH * 2)     //  66048
#define LDS_TOTAL     (LDS_QKV_BYTES + LDS_X_BYTES) // 263168 (<320KB WGP LDS)

// =====================================================================
// Kernel 0: convert fp32 weights to transposed bf16 layouts in workspace.
//   wqkv_t[n][k], n in [0,3072): n -> (matrix m = n/1024, head h, depth d)
//   wf_t  [n][k] = Wf[k][n]
// K-contiguous rows => WMMA B-fragment loads are two global b128 per lane.
// =====================================================================
__global__ void convert_weights_kernel(const float* __restrict__ Wq,
                                       const float* __restrict__ Wk,
                                       const float* __restrict__ Wv,
                                       const float* __restrict__ Wf,
                                       bf16_t* __restrict__ wqkv_t,
                                       bf16_t* __restrict__ wf_t) {
    int idx = blockIdx.x * blockDim.x + threadIdx.x;
    const int total_qkv = QKV_N * F_DIM;
    if (idx < total_qkv) {
        int n = idx >> 10;           // / F_DIM
        int k = idx & (F_DIM - 1);
        int m = n >> 10;             // 0=q, 1=k, 2=v
        int r = n & 1023;
        int h = r >> 6;
        int d = r & 63;
        const float* W = (m == 0) ? Wq : ((m == 1) ? Wk : Wv);
        float v = W[(h << 16) + (k << 6) + d];   // [H][F][D]
        wqkv_t[idx] = (bf16_t)v;
    } else {
        int j = idx - total_qkv;
        if (j < F_DIM * OUT_DIM) {
            int n = j >> 10;
            int k = j & (F_DIM - 1);
            wf_t[j] = (bf16_t)Wf[k * OUT_DIM + n];
        }
    }
}

// =====================================================================
// Kernel 1: fully fused MHSA for a 32-row tile, 8 waves / 256 threads.
//   Phase 0: x tile fp32 -> bf16 LDS
//   Phase 1: QKV = x @ Wqkv      (WMMA bf16, acc fp32) -> LDS
//   Phase 2: per-row 16x16 head attention + softmax     (VALU) -> concat LDS
//   Phase 3: out = concat @ Wf + bias (WMMA bf16, acc fp32) -> global
// =====================================================================
__launch_bounds__(256, 1)
__global__ void fused_mhsa_kernel(const float* __restrict__ x,
                                  const bf16_t* __restrict__ wqkv_t,
                                  const bf16_t* __restrict__ wf_t,
                                  const float* __restrict__ bias,
                                  float* __restrict__ out) {
    extern __shared__ char smem[];
    bf16_t* sQKV = (bf16_t*)smem;                      // [32][QKV_PITCH]  q|k|v
    bf16_t* sX   = (bf16_t*)(smem + LDS_QKV_BYTES);    // [32][X_PITCH]    x, later concat

    const int tid    = threadIdx.x;
    const int lane   = tid & 31;          // wave32
    const int wave   = tid >> 5;          // 8 waves
    const int laneLo = lane & 15;
    const int laneHi = lane >> 4;         // 0 / 1
    const int rowBase = blockIdx.x * TILE_ROWS;

    // ---------------- Phase 0: stage x tile as bf16 ----------------
    {
        const float* xrow = x + (size_t)rowBase * F_DIM;
        for (int i = tid; i < TILE_ROWS * (F_DIM / 4); i += 256) {
            int r = i >> 8;                 // / 256 float4-per-row
            int c = (i & 255) << 2;
            float4 v = *(const float4*)(xrow + (size_t)r * F_DIM + c);
            bf16x4 p = { (bf16_t)v.x, (bf16_t)v.y, (bf16_t)v.z, (bf16_t)v.w };
            *(bf16x4*)(sX + r * X_PITCH + c) = p;       // ds_store_b64
        }
    }
    __syncthreads();

    // ---------------- Phase 1: QKV projection GEMM ----------------
    // Each wave owns a strided set of 16-wide N-tiles; 2 M-tiles share B frag.
    {
        const int koff = laneHi * 8;    // A-frag K sub-offset per lane half
        for (int nt = wave; nt < QKV_N / 16; nt += 8) {
            f32x8 acc0 = {};
            f32x8 acc1 = {};
            const int bcol = nt * 16 + laneLo;
            const bf16_t* bptr = wqkv_t + (size_t)bcol * F_DIM + laneHi * 8;
            for (int kc = 0; kc < F_DIM; kc += 32) {
                bf16x16 bfrag, af0, af1;
                ((uint4*)&bfrag)[0] = *(const uint4*)(bptr + kc);       // K .. K+7
                ((uint4*)&bfrag)[1] = *(const uint4*)(bptr + kc + 16);  // K+16 .. K+23 (lane half)
                const bf16_t* a0 = sX + laneLo * X_PITCH + kc + koff;
                const bf16_t* a1 = sX + (16 + laneLo) * X_PITCH + kc + koff;
                ((uint4*)&af0)[0] = *(const uint4*)(a0);
                ((uint4*)&af0)[1] = *(const uint4*)(a0 + 16);
                ((uint4*)&af1)[0] = *(const uint4*)(a1);
                ((uint4*)&af1)[1] = *(const uint4*)(a1 + 16);
                acc0 = __builtin_amdgcn_wmma_f32_16x16x32_bf16(
                           false, af0, false, bfrag, (short)0, acc0, false, false);
                acc1 = __builtin_amdgcn_wmma_f32_16x16x32_bf16(
                           false, af1, false, bfrag, (short)0, acc1, false, false);
            }
            // C layout: VGPR i, lanes 0-15 -> M=i ; lanes 16-31 -> M=i+8
            const int col = nt * 16 + laneLo;
#pragma unroll
            for (int i = 0; i < 8; ++i) {
                int r0 = i + laneHi * 8;
                sQKV[r0 * QKV_PITCH + col]        = (bf16_t)acc0[i];
                sQKV[(16 + r0) * QKV_PITCH + col] = (bf16_t)acc1[i];
            }
        }
    }
    __syncthreads();

    // ---------------- Phase 2: per-row head attention (VALU) ----------------
    // scores[h,g] = (q_h . k_g) / 8 ; softmax over g (16) ; o = w @ v
    {
        for (int task = tid; task < TILE_ROWS * H_NUM; task += 256) {
            const int row = task >> 4;
            const int h   = task & 15;
            const bf16_t* base = sQKV + row * QKV_PITCH;
            const bf16_t* qv = base + h * D_DIM;           // q_h
            float s[H_NUM];
            float smax = -3.0e38f;
#pragma unroll
            for (int g = 0; g < H_NUM; ++g) {
                const bf16_t* kv = base + 1024 + g * D_DIM;
                float acc = 0.f;
#pragma unroll
                for (int d = 0; d < D_DIM; ++d)
                    acc += (float)qv[d] * (float)kv[d];
                acc *= 0.125f;                             // 1/sqrt(64)
                s[g] = acc;
                smax = fmaxf(smax, acc);
            }
            float ssum = 0.f;
#pragma unroll
            for (int g = 0; g < H_NUM; ++g) { s[g] = __expf(s[g] - smax); ssum += s[g]; }
            const float inv = 1.f / ssum;
#pragma unroll
            for (int g = 0; g < H_NUM; ++g) s[g] *= inv;

            // o in chunks of 8 to keep VGPR pressure low; write transposed concat
            bf16_t* crow = sX + row * X_PITCH + h;         // concat[row][d*16 + h]
#pragma unroll
            for (int dc = 0; dc < D_DIM; dc += 8) {
                float oc[8] = {0.f, 0.f, 0.f, 0.f, 0.f, 0.f, 0.f, 0.f};
#pragma unroll
                for (int g = 0; g < H_NUM; ++g) {
                    const bf16_t* vv = base + 2048 + g * D_DIM + dc;
                    const float wg = s[g];
#pragma unroll
                    for (int j = 0; j < 8; ++j) oc[j] += wg * (float)vv[j];
                }
#pragma unroll
                for (int j = 0; j < 8; ++j) crow[(dc + j) * 16] = (bf16_t)oc[j];
            }
        }
    }
    __syncthreads();

    // ---------------- Phase 3: final GEMM  out = concat @ Wf + bias ----------------
    {
        const int koff = laneHi * 8;
        for (int nt = wave; nt < OUT_DIM / 16; nt += 8) {
            f32x8 acc0 = {};
            f32x8 acc1 = {};
            const int bcol = nt * 16 + laneLo;
            const bf16_t* bptr = wf_t + (size_t)bcol * F_DIM + laneHi * 8;
            for (int kc = 0; kc < F_DIM; kc += 32) {
                bf16x16 bfrag, af0, af1;
                ((uint4*)&bfrag)[0] = *(const uint4*)(bptr + kc);
                ((uint4*)&bfrag)[1] = *(const uint4*)(bptr + kc + 16);
                const bf16_t* a0 = sX + laneLo * X_PITCH + kc + koff;
                const bf16_t* a1 = sX + (16 + laneLo) * X_PITCH + kc + koff;
                ((uint4*)&af0)[0] = *(const uint4*)(a0);
                ((uint4*)&af0)[1] = *(const uint4*)(a0 + 16);
                ((uint4*)&af1)[0] = *(const uint4*)(a1);
                ((uint4*)&af1)[1] = *(const uint4*)(a1 + 16);
                acc0 = __builtin_amdgcn_wmma_f32_16x16x32_bf16(
                           false, af0, false, bfrag, (short)0, acc0, false, false);
                acc1 = __builtin_amdgcn_wmma_f32_16x16x32_bf16(
                           false, af1, false, bfrag, (short)0, acc1, false, false);
            }
            const int col = nt * 16 + laneLo;
            const float bv = bias[col];
#pragma unroll
            for (int i = 0; i < 8; ++i) {
                int r0 = i + laneHi * 8;
                out[(size_t)(rowBase + r0) * OUT_DIM + col]      = acc0[i] + bv;
                out[(size_t)(rowBase + 16 + r0) * OUT_DIM + col] = acc1[i] + bv;
            }
        }
    }
}

// =====================================================================
// host launcher
// =====================================================================
extern "C" void kernel_launch(void* const* d_in, const int* in_sizes, int n_in,
                              void* d_out, int out_size, void* d_ws, size_t ws_size,
                              hipStream_t stream) {
    const float* x  = (const float*)d_in[0];
    const float* Wq = (const float*)d_in[1];
    const float* Wk = (const float*)d_in[2];
    const float* Wv = (const float*)d_in[3];
    const float* Wf = (const float*)d_in[4];
    const float* bf = (const float*)d_in[5];
    float* out = (float*)d_out;

    // workspace: bf16 transposed weights (6MB + 2MB)
    bf16_t* wqkv_t = (bf16_t*)d_ws;
    bf16_t* wf_t   = (bf16_t*)((char*)d_ws + (size_t)QKV_N * F_DIM * sizeof(bf16_t));

    const int totalConv = QKV_N * F_DIM + F_DIM * OUT_DIM;   // 4,194,304
    convert_weights_kernel<<<(totalConv + 255) / 256, 256, 0, stream>>>(
        Wq, Wk, Wv, Wf, wqkv_t, wf_t);

    (void)hipFuncSetAttribute((const void*)fused_mhsa_kernel,
                              hipFuncAttributeMaxDynamicSharedMemorySize, LDS_TOTAL);
    fused_mhsa_kernel<<<NROWS / TILE_ROWS, 256, LDS_TOTAL, stream>>>(
        x, wqkv_t, wf_t, bf, out);
}
